// SlotAttentionGroupingIterCA_58076547776888
// MI455X (gfx1250) — compile-verified, matrix operand order checked by reading, not run on hardware
//
#include <hip/hip_runtime.h>
#include <hip/hip_bf16.h>
#include <stdint.h>

// ---------------------------------------------------------------------------
// SlotAttention + VQ + cross-attention feedback for MI455X (gfx1250, wave32).
// All large GEMMs run through one batched bf16 WMMA GEMM kernel
// (v_wmma_f32_16x16x32_bf16, f32 accumulation). 128x64 block tile, 8 waves,
// each wave owns a 2x2 grid of 16x16 C tiles (4 WMMA per K-step). Tiles are
// staged global->LDS with GLOBAL_LOAD_ASYNC_TO_LDS_B128 when available
// (ASYNCcnt / s_wait_asynccnt), else via register staging.
// ---------------------------------------------------------------------------

typedef __bf16 bf16_t;
typedef __attribute__((ext_vector_type(16))) __bf16 v16bf;
typedef __attribute__((ext_vector_type(8)))  float  v8f;
typedef __attribute__((ext_vector_type(4)))  int    v4i_;

#ifndef __has_builtin
#define __has_builtin(x) 0
#endif
#if __has_builtin(__builtin_amdgcn_global_load_async_to_lds_b128)
#define HAVE_ASYNC_LDS 1
#else
#define HAVE_ASYNC_LDS 0
#endif

#define B_    32
#define N_    1024
#define DF_   768
#define DO_   256
#define KVQ_  256
#define H_    4
#define S_    16
#define KC_   512
#define CAH_  4
#define ITERS_ 3
#define DPH_  64
#define CAD_  192

enum { GF_BIAS = 1, GF_RELU = 2, GF_ACCUM = 4 };

// 16-byte global -> LDS staging (async DMA path on CDNA5 if present).
__device__ __forceinline__ void stage16(const bf16_t* g, bf16_t* l)
{
#if HAVE_ASYNC_LDS
    __builtin_amdgcn_global_load_async_to_lds_b128(
        (__attribute__((address_space(1))) v4i_*)(uintptr_t)g,
        (__attribute__((address_space(3))) v4i_*)(uint32_t)(uintptr_t)l,
        0, 0);
#else
    *(uint4*)l = *(const uint4*)g;
#endif
}

__device__ __forceinline__ void wait_stage()
{
#if HAVE_ASYNC_LDS
#if __has_builtin(__builtin_amdgcn_s_wait_asynccnt)
    __builtin_amdgcn_s_wait_asynccnt(0);
#else
    asm volatile("s_wait_asynccnt 0" ::: "memory");
#endif
#endif
}

// ---------------------------------------------------------------------------
// Generic batched NT GEMM: C[z] = A[z] (M x K) * B[z] (N x K)^T, bf16 -> f32.
// z decomposed as (zb, zh) = (z / zdiv, z % zdiv) with independent strides.
// Block tile 128(M) x 64(N), K stepped by 32, double-buffered LDS.
// OOB tile rows are clamped (not branched): junk lands only in C rows/cols
// that the guarded epilogue never stores.
// ---------------------------------------------------------------------------
__global__ __launch_bounds__(256) void gemm_bf16_nt(
    const bf16_t* __restrict__ A, long long sAb, long long sAh, int lda,
    const bf16_t* __restrict__ Bm, long long sBb, long long sBh, int ldb,
    float* __restrict__ Cf, bf16_t* __restrict__ Cb,
    long long sCb, long long sCh, int ldc,
    int M, int N, int K, int zdiv,
    const float* __restrict__ bias, int flags)
{
    __shared__ bf16_t As[2][128 * 40];   // 40-elem row stride: 80B, 16B aligned
    __shared__ bf16_t Bs[2][64 * 40];

    const int z  = blockIdx.z;
    const int zb = z / zdiv, zh = z % zdiv;
    const bf16_t* Ab = A  + (long long)zb * sAb + (long long)zh * sAh;
    const bf16_t* Bb = Bm + (long long)zb * sBb + (long long)zh * sBh;
    const long long coff = (long long)zb * sCb + (long long)zh * sCh;

    const int t    = threadIdx.x;
    const int bm0  = blockIdx.x * 128;
    const int bn0  = blockIdx.y * 64;

    const int lane  = t & 31;
    const int wv    = t >> 5;
    const int mw    = wv & 3;          // 4 M-strips of 32 rows
    const int nw    = wv >> 2;         // 2 N-strips of 32 cols
    const int rc    = lane & 15;
    const int khalf = (lane & 16) ? 8 : 0;   // per-ISA 16-bit A/B lane layout

    v8f acc[2][2];
    #pragma unroll
    for (int i = 0; i < 2; ++i)
        #pragma unroll
        for (int j = 0; j < 2; ++j)
            acc[i][j] = (v8f){0.f,0.f,0.f,0.f,0.f,0.f,0.f,0.f};

    const int nk = K >> 5;

    auto loadTiles = [&](int buf, int k0) {
        #pragma unroll
        for (int i = 0; i < 2; ++i) {                 // A: 128 rows x 32 cols
            const int li  = t + (i << 8);
            const int row = li >> 2;
            const int kg  = (li & 3) << 3;
            int gm = bm0 + row; if (gm >= M) gm = M - 1;
            stage16(Ab + (long long)gm * lda + k0 + kg, &As[buf][row * 40 + kg]);
        }
        {                                             // B: 64 rows x 32 cols
            const int row = t >> 2;
            const int kg  = (t & 3) << 3;
            int gn = bn0 + row; if (gn >= N) gn = N - 1;
            stage16(Bb + (long long)gn * ldb + k0 + kg, &Bs[buf][row * 40 + kg]);
        }
    };

    loadTiles(0, 0);
    wait_stage();
    __syncthreads();
    int buf = 0;
    for (int kt = 0; kt < nk; ++kt) {
        if (kt + 1 < nk) loadTiles(buf ^ 1, (kt + 1) << 5);
        union { v16bf v; uint4 q[2]; } a0, a1, b0, b1;
        const int ar0 = mw * 32 + rc;
        a0.q[0] = *(const uint4*)(&As[buf][ar0 * 40 + khalf]);
        a0.q[1] = *(const uint4*)(&As[buf][ar0 * 40 + khalf + 16]);
        const int ar1 = ar0 + 16;
        a1.q[0] = *(const uint4*)(&As[buf][ar1 * 40 + khalf]);
        a1.q[1] = *(const uint4*)(&As[buf][ar1 * 40 + khalf + 16]);
        const int br0 = nw * 32 + rc;
        b0.q[0] = *(const uint4*)(&Bs[buf][br0 * 40 + khalf]);
        b0.q[1] = *(const uint4*)(&Bs[buf][br0 * 40 + khalf + 16]);
        const int br1 = br0 + 16;
        b1.q[0] = *(const uint4*)(&Bs[buf][br1 * 40 + khalf]);
        b1.q[1] = *(const uint4*)(&Bs[buf][br1 * 40 + khalf + 16]);
        acc[0][0] = __builtin_amdgcn_wmma_f32_16x16x32_bf16(false, a0.v, false, b0.v,
                                                            (short)0, acc[0][0], false, false);
        acc[0][1] = __builtin_amdgcn_wmma_f32_16x16x32_bf16(false, a0.v, false, b1.v,
                                                            (short)0, acc[0][1], false, false);
        acc[1][0] = __builtin_amdgcn_wmma_f32_16x16x32_bf16(false, a1.v, false, b0.v,
                                                            (short)0, acc[1][0], false, false);
        acc[1][1] = __builtin_amdgcn_wmma_f32_16x16x32_bf16(false, a1.v, false, b1.v,
                                                            (short)0, acc[1][1], false, false);
        wait_stage();
        __syncthreads();
        buf ^= 1;
    }

    // Epilogue per documented C layout: VGPR e, lanes<16 -> M=e, lanes>=16 -> M=e+8.
    const int roff = (lane & 16) ? 8 : 0;
    float biasv[2] = {0.f, 0.f};
    if (flags & GF_BIAS) {
        #pragma unroll
        for (int tj = 0; tj < 2; ++tj) {
            const int gn = bn0 + nw * 32 + tj * 16 + rc;
            if (gn < N) biasv[tj] = bias[gn];
        }
    }
    #pragma unroll
    for (int ti = 0; ti < 2; ++ti) {
        #pragma unroll
        for (int e = 0; e < 8; ++e) {
            const int gm = bm0 + mw * 32 + ti * 16 + roff + e;
            if (gm >= M) continue;
            const long long rb = coff + (long long)gm * ldc;
            #pragma unroll
            for (int tj = 0; tj < 2; ++tj) {
                const int gn = bn0 + nw * 32 + tj * 16 + rc;
                if (gn >= N) continue;
                float x = acc[ti][tj][e] + biasv[tj];
                if (flags & GF_RELU) x = fmaxf(x, 0.f);
                if (Cf) { if (flags & GF_ACCUM) Cf[rb + gn] += x; else Cf[rb + gn] = x; }
                if (Cb) Cb[rb + gn] = (bf16_t)x;
            }
        }
    }
}

// ---------------------------------------------------------------------------
// Weight prep: f32 [R,C] -> bf16 [C,R] (NT-ready), plain convert, code norms.
// ---------------------------------------------------------------------------
__global__ void wt_transpose_kernel(const float* __restrict__ in,
                                    bf16_t* __restrict__ out, int R, int C)
{
    int idx = blockIdx.x * 256 + threadIdx.x;
    if (idx >= R * C) return;
    int r = idx % R;
    int c = idx / R;
    out[idx] = (bf16_t)in[(long long)r * C + c];
}

__global__ void f32_to_bf16_kernel(const float* __restrict__ in,
                                   bf16_t* __restrict__ out, int n)
{
    int idx = blockIdx.x * 256 + threadIdx.x;
    if (idx < n) out[idx] = (bf16_t)in[idx];
}

__global__ void cnorm_kernel(const float* __restrict__ cb, float* __restrict__ cn)
{
    int c = blockIdx.x * 256 + threadIdx.x;
    if (c >= KC_) return;
    float s = 0.f;
    for (int i = 0; i < DO_; ++i) { float v = cb[(long long)c * DO_ + i]; s += v * v; }
    cn[c] = s;
}

// ---------------------------------------------------------------------------
// (inputs [+ positions@Wpos + bpos]) -> LayerNorm -> bf16. One block per token.
// ---------------------------------------------------------------------------
__global__ __launch_bounds__(256) void posln_kernel(
    const float* __restrict__ in, const float* __restrict__ pos,
    const float* __restrict__ Wpos, const float* __restrict__ bpos,
    const float* __restrict__ g, const float* __restrict__ bb,
    bf16_t* __restrict__ out)
{
    const int tok = blockIdx.x;
    const int t = threadIdx.x;
    __shared__ float red[256];
    float p0 = 0.f, p1 = 0.f, p2 = 0.f, p3 = 0.f;
    if (pos) { const float* pp = pos + (long long)tok * 4; p0 = pp[0]; p1 = pp[1]; p2 = pp[2]; p3 = pp[3]; }
    float vals[3];
    float s = 0.f;
    #pragma unroll
    for (int i = 0; i < 3; ++i) {
        int f = t + i * 256;
        float v = in[(long long)tok * DF_ + f];
        if (pos) v += p0 * Wpos[f] + p1 * Wpos[DF_ + f] + p2 * Wpos[2 * DF_ + f]
                    + p3 * Wpos[3 * DF_ + f] + bpos[f];
        vals[i] = v; s += v;
    }
    red[t] = s; __syncthreads();
    for (int o = 128; o > 0; o >>= 1) { if (t < o) red[t] += red[t + o]; __syncthreads(); }
    const float mean = red[0] * (1.f / DF_);
    __syncthreads();
    float s2 = 0.f;
    #pragma unroll
    for (int i = 0; i < 3; ++i) { float d = vals[i] - mean; s2 += d * d; }
    red[t] = s2; __syncthreads();
    for (int o = 128; o > 0; o >>= 1) { if (t < o) red[t] += red[t + o]; __syncthreads(); }
    const float rstd = rsqrtf(red[0] * (1.f / DF_) + 1e-5f);
    #pragma unroll
    for (int i = 0; i < 3; ++i) {
        int f = t + i * 256;
        out[(long long)tok * DF_ + f] = (bf16_t)((vals[i] - mean) * rstd * g[f] + bb[f]);
    }
}

// LayerNorm over rows of 256 -> bf16. One block per row.
__global__ __launch_bounds__(256) void ln256_kernel(
    const float* __restrict__ in, const float* __restrict__ g,
    const float* __restrict__ bb, bf16_t* __restrict__ out)
{
    const int r = blockIdx.x;
    const int t = threadIdx.x;
    __shared__ float red[256];
    const float v = in[(long long)r * 256 + t];
    red[t] = v; __syncthreads();
    for (int o = 128; o > 0; o >>= 1) { if (t < o) red[t] += red[t + o]; __syncthreads(); }
    const float mean = red[0] * (1.f / 256.f);
    __syncthreads();
    const float d = v - mean;
    red[t] = d * d; __syncthreads();
    for (int o = 128; o > 0; o >>= 1) { if (t < o) red[t] += red[t + o]; __syncthreads(); }
    const float rstd = rsqrtf(red[0] * (1.f / 256.f) + 1e-5f);
    out[(long long)r * 256 + t] = (bf16_t)(d * rstd * g[t] + bb[t]);
}

// v bf16 [B,1024,256] -> vT bf16 [B,256,1024] tiled transpose.
__global__ __launch_bounds__(256) void transpose_v_kernel(
    const bf16_t* __restrict__ in, bf16_t* __restrict__ out)
{
    __shared__ bf16_t tile[32][33];
    const int b  = blockIdx.z;
    const int n0 = blockIdx.x * 32;
    const int c0 = blockIdx.y * 32;
    const int tx = threadIdx.x & 31;
    const int ty = threadIdx.x >> 5;
    const bf16_t* src = in + (long long)b * N_ * KVQ_;
    bf16_t* dst = out + (long long)b * KVQ_ * N_;
    #pragma unroll
    for (int j = 0; j < 4; ++j)
        tile[ty + j * 8][tx] = src[(long long)(n0 + ty + j * 8) * KVQ_ + c0 + tx];
    __syncthreads();
    #pragma unroll
    for (int j = 0; j < 4; ++j)
        dst[(long long)(c0 + ty + j * 8) * N_ + n0 + tx] = tile[tx][ty + j * 8];
}

// Softmax over the 64 slot-heads (axis S*H) per (b,n); also head-mean attn out.
__global__ __launch_bounds__(256) void softmax_sh_kernel(
    const float* __restrict__ dots, float* __restrict__ attnf,
    float* __restrict__ attn_out)
{
    const int idx = blockIdx.x * 256 + threadIdx.x;   // b*N + n
    const int b = idx >> 10;
    const int n = idx & 1023;
    const long long base = (long long)b * (S_ * H_) * N_ + n;
    float v[64];
    float mx = -3.4e38f;
    #pragma unroll
    for (int sh = 0; sh < 64; ++sh) {
        float x = dots[base + (long long)sh * N_] * 0.125f;   // DPH^-0.5
        v[sh] = x; mx = fmaxf(mx, x);
    }
    float sum = 0.f;
    #pragma unroll
    for (int sh = 0; sh < 64; ++sh) { float e = __expf(v[sh] - mx); v[sh] = e; sum += e; }
    const float inv = 1.f / sum;
    float asum[16];
    #pragma unroll
    for (int s = 0; s < 16; ++s) asum[s] = 0.f;
    #pragma unroll
    for (int sh = 0; sh < 64; ++sh) {
        float a = v[sh] * inv;
        attnf[base + (long long)sh * N_] = a;
        asum[sh >> 2] += a;
    }
    #pragma unroll
    for (int s = 0; s < 16; ++s)
        attn_out[((long long)b * 16 + s) * N_ + n] = asum[s] * 0.25f;
}

// (attn + EPS) / sum_n(attn + EPS) -> bf16. One block per (b,s,h).
__global__ __launch_bounds__(256) void renorm_kernel(
    const float* __restrict__ attnf, bf16_t* __restrict__ out)
{
    const long long base = (long long)blockIdx.x * N_;
    const int t = threadIdx.x;
    __shared__ float red[256];
    float s = 0.f;
    #pragma unroll
    for (int i = 0; i < 4; ++i) s += attnf[base + t + i * 256];
    red[t] = s; __syncthreads();
    for (int o = 128; o > 0; o >>= 1) { if (t < o) red[t] += red[t + o]; __syncthreads(); }
    const float inv = 1.f / (red[0] + 1024.f * 1e-8f);
    #pragma unroll
    for (int i = 0; i < 4; ++i) {
        int n = t + i * 256;
        out[base + n] = (bf16_t)((attnf[base + n] + 1e-8f) * inv);
    }
}

// Fused GRU gates; in-place slots update (reads hprev, writes new slots).
__global__ void gru_kernel(const float* __restrict__ gi,
                           const float* __restrict__ gh,
                           float* __restrict__ slots)
{
    const int idx = blockIdx.x * 256 + threadIdx.x;   // < 512*256
    const int r = idx >> 8, c = idx & 255;
    const float* gir = gi + (long long)r * 768;
    const float* ghr = gh + (long long)r * 768;
    const float rr = 1.f / (1.f + __expf(-(gir[c] + ghr[c])));
    const float zz = 1.f / (1.f + __expf(-(gir[256 + c] + ghr[256 + c])));
    const float ng = tanhf(gir[512 + c] + rr * ghr[512 + c]);
    const float h  = slots[idx];
    slots[idx] = (1.f - zz) * ng + zz * h;
}

// VQ: argmin over codes, gather quantized slots (bf16), commit-loss partial sum.
__global__ __launch_bounds__(256) void vq_kernel(
    const float* __restrict__ dp, const float* __restrict__ cn,
    const float* __restrict__ slots, const float* __restrict__ cb,
    bf16_t* __restrict__ qsl, int* __restrict__ inds,
    float* __restrict__ out_inds, float* __restrict__ acc)
{
    const int r = blockIdx.x;
    const int t = threadIdx.x;
    __shared__ float sv[256];
    __shared__ int   si[256];
    float best = 3.4e38f; int bi = 0;
    for (int c = t; c < KC_; c += 256) {
        float d = cn[c] - 2.f * dp[(long long)r * KC_ + c];
        if (d < best) { best = d; bi = c; }
    }
    sv[t] = best; si[t] = bi; __syncthreads();
    for (int o = 128; o > 0; o >>= 1) {
        if (t < o) {
            if (sv[t + o] < sv[t] || (sv[t + o] == sv[t] && si[t + o] < si[t])) {
                sv[t] = sv[t + o]; si[t] = si[t + o];
            }
        }
        __syncthreads();
    }
    const int ind = si[0];
    if (t == 0) { inds[r] = ind; out_inds[r] = (float)ind; }
    const float cv = cb[(long long)ind * DO_ + t];
    qsl[(long long)r * DO_ + t] = (bf16_t)cv;
    const float d = slots[(long long)r * DO_ + t] - cv;
    __syncthreads();
    sv[t] = d * d; __syncthreads();
    for (int o = 128; o > 0; o >>= 1) { if (t < o) sv[t] += sv[t + o]; __syncthreads(); }
    if (t == 0) atomicAdd(acc, sv[0]);
}

// CA softmax over S=16 per (b,h,n); writes K=32-padded bf16 weights + ca_attn.
__global__ __launch_bounds__(256) void ca_softmax_kernel(
    const float* __restrict__ logits, bf16_t* __restrict__ w,
    float* __restrict__ ca_out)
{
    const int idx = blockIdx.x * 256 + threadIdx.x;   // b*N + n
    const int b = idx >> 10, n = idx & 1023;
    const float rs = 0.0721687836f;                   // 1/sqrt(192)
    float ca[16];
    #pragma unroll
    for (int s = 0; s < 16; ++s) ca[s] = 0.f;
    for (int h = 0; h < 4; ++h) {
        const long long lb = (((long long)b * 4 + h) * 1024 + n) * 16;
        float l[16]; float mx = -3.4e38f;
        #pragma unroll
        for (int k = 0; k < 16; ++k) { float x = logits[lb + k] * rs; l[k] = x; mx = fmaxf(mx, x); }
        float sum = 0.f;
        #pragma unroll
        for (int k = 0; k < 16; ++k) { float e = __expf(l[k] - mx); l[k] = e; sum += e; }
        const float inv = 1.f / sum;
        const long long wb = (((long long)b * 4 + h) * 1024 + n) * 32;
        #pragma unroll
        for (int k = 0; k < 16; ++k) { float wv = l[k] * inv; w[wb + k] = (bf16_t)wv; ca[k] += wv; }
        #pragma unroll
        for (int k = 16; k < 32; ++k) w[wb + k] = (bf16_t)0.f;
    }
    #pragma unroll
    for (int s = 0; s < 16; ++s)
        ca_out[((long long)b * 16 + s) * 1024 + n] = ca[s] * 0.25f;
}

// vf bf16 [B,16,768] -> vfT bf16 [B,H,192,32] (K padded 16 -> 32 with zeros).
__global__ void vft_kernel(const bf16_t* __restrict__ vf, bf16_t* __restrict__ vfT)
{
    const int idx = blockIdx.x * 256 + threadIdx.x;   // < B*H*192*32
    const int k    = idx & 31;
    const int rest = idx >> 5;
    const int d  = rest % 192;
    const int bh = rest / 192;
    const int h  = bh & 3;
    const int b  = bh >> 2;
    bf16_t v = (bf16_t)0.f;
    if (k < 16) v = vf[((long long)(b * 16 + k)) * 768 + h * 192 + d];
    vfT[idx] = v;
}

__global__ void init_acc_kernel(float* acc) { if (threadIdx.x == 0) acc[0] = 0.f; }
__global__ void commit_final_kernel(const float* acc, float* out)
{
    if (threadIdx.x == 0) out[0] = acc[0] / (float)(ITERS_ * B_ * S_ * DO_);
}

// ---------------------------------------------------------------------------
// Host side
// ---------------------------------------------------------------------------
static void gemm(hipStream_t st,
                 const bf16_t* A, long long sAb, long long sAh, int lda,
                 const bf16_t* Bm, long long sBb, long long sBh, int ldb,
                 float* Cf, bf16_t* Cb, long long sCb, long long sCh, int ldc,
                 int M, int N, int K, int Z, int zdiv,
                 const float* bias, int flags)
{
    dim3 g((unsigned)((M + 127) / 128), (unsigned)((N + 63) / 64), (unsigned)Z);
    gemm_bf16_nt<<<g, 256, 0, st>>>(A, sAb, sAh, lda, Bm, sBb, sBh, ldb,
                                    Cf, Cb, sCb, sCh, ldc, M, N, K, zdiv, bias, flags);
}

extern "C" void kernel_launch(void* const* d_in, const int* in_sizes, int n_in,
                              void* d_out, int out_size, void* d_ws, size_t ws_size,
                              hipStream_t stream)
{
    (void)in_sizes; (void)n_in; (void)out_size; (void)ws_size;
    const float* features     = (const float*)d_in[0];
    const float* positions    = (const float*)d_in[1];
    const float* conditioning = (const float*)d_in[2];
    const float* Wpos   = (const float*)d_in[3];
    const float* bpos   = (const float*)d_in[4];
    const float* ln_in_g = (const float*)d_in[5];
    const float* ln_in_b = (const float*)d_in[6];
    const float* ln_sl_g = (const float*)d_in[7];
    const float* ln_sl_b = (const float*)d_in[8];
    const float* ln_ft_g = (const float*)d_in[9];
    const float* ln_ft_b = (const float*)d_in[10];
    const float* Wk = (const float*)d_in[11];
    const float* Wv = (const float*)d_in[12];
    const float* Wq = (const float*)d_in[13];
    const float* gru_Wih = (const float*)d_in[14];
    const float* gru_Whh = (const float*)d_in[15];
    const float* gru_bih = (const float*)d_in[16];
    const float* gru_bhh = (const float*)d_in[17];
    const float* mlp_ln_g = (const float*)d_in[18];
    const float* mlp_ln_b = (const float*)d_in[19];
    const float* mlp_W1 = (const float*)d_in[20];
    const float* mlp_b1 = (const float*)d_in[21];
    const float* mlp_W2 = (const float*)d_in[22];
    const float* mlp_b2 = (const float*)d_in[23];
    const float* codebook = (const float*)d_in[24];
    const float* ca_Wq = (const float*)d_in[25];
    const float* ca_bq = (const float*)d_in[26];
    const float* ca_Wk = (const float*)d_in[27];
    const float* ca_bk = (const float*)d_in[28];
    const float* ca_Wv = (const float*)d_in[29];
    const float* ca_bv = (const float*)d_in[30];
    const float* ca_Wo = (const float*)d_in[31];
    const float* ca_bo = (const float*)d_in[32];

    float* out = (float*)d_out;
    const size_t OUT_SLOTS  = 0;
    const size_t OUT_INPUTS = OUT_SLOTS + (size_t)B_ * S_ * DO_;
    const size_t OUT_ATTN   = OUT_INPUTS + (size_t)B_ * N_ * DF_;
    const size_t OUT_CA     = OUT_ATTN + (size_t)B_ * S_ * N_;
    const size_t OUT_COMMIT = OUT_CA + (size_t)B_ * S_ * N_;
    const size_t OUT_INDS   = OUT_COMMIT + 1;

    // ---- carve workspace -------------------------------------------------
    char* base = (char*)d_ws; size_t off = 0;
    auto alloc = [&](size_t bytes) -> void* {
        void* r = base + off; off = (off + bytes + 255) & ~(size_t)255; return r;
    };
    const size_t TOK = (size_t)B_ * N_;             // 32768
    float*  inputs  = (float*)alloc(TOK * DF_ * 4);
    bf16_t* kinbuf  = (bf16_t*)alloc(TOK * DF_ * 2);    // kin / lnft / ctx (reused)
    bf16_t* qf      = (bf16_t*)alloc(TOK * DF_ * 2);
    bf16_t* kbuf    = (bf16_t*)alloc(TOK * KVQ_ * 2);
    bf16_t* vbuf    = (bf16_t*)alloc(TOK * KVQ_ * 2);
    bf16_t* vT      = (bf16_t*)alloc(TOK * KVQ_ * 2);
    float*  dots    = (float*)alloc((size_t)B_ * S_ * H_ * N_ * 4);  // also logits
    void*   afw     = alloc((size_t)B_ * S_ * H_ * N_ * 4);          // attnf f32 / w bf16
    float*  attnf   = (float*)afw;
    bf16_t* wbuf    = (bf16_t*)afw;
    bf16_t* attnb   = (bf16_t*)alloc((size_t)B_ * S_ * H_ * N_ * 2);
    float*  slots   = (float*)alloc((size_t)B_ * S_ * DO_ * 4);
    bf16_t* snbf    = (bf16_t*)alloc(512 * 256 * 2);
    bf16_t* tbf     = (bf16_t*)alloc(512 * 256 * 2);
    bf16_t* slotsbf = (bf16_t*)alloc(512 * 256 * 2);
    bf16_t* qsl     = (bf16_t*)alloc(512 * 256 * 2);
    bf16_t* updates = (bf16_t*)alloc(512 * 256 * 2);
    bf16_t* qbf     = (bf16_t*)alloc(512 * 256 * 2);
    float*  gi      = (float*)alloc(512 * 768 * 4);
    float*  gh      = (float*)alloc(512 * 768 * 4);
    bf16_t* h1bf    = (bf16_t*)alloc(512 * 1024 * 2);
    float*  dp      = (float*)alloc(512 * 512 * 4);
    bf16_t* kf      = (bf16_t*)alloc(512 * 768 * 2);
    bf16_t* vfb     = (bf16_t*)alloc(512 * 768 * 2);
    bf16_t* vfT     = (bf16_t*)alloc((size_t)B_ * H_ * CAD_ * 32 * 2);
    float*  cn      = (float*)alloc(KC_ * 4);
    int*    indsbuf = (int*)alloc(KC_ * 4);
    float*  acc     = (float*)alloc(256);
    bf16_t* WkT   = (bf16_t*)alloc((size_t)DF_ * KVQ_ * 2);
    bf16_t* WvT   = (bf16_t*)alloc((size_t)DF_ * KVQ_ * 2);
    bf16_t* WqT   = (bf16_t*)alloc((size_t)DO_ * KVQ_ * 2);
    bf16_t* WihT  = (bf16_t*)alloc((size_t)KVQ_ * 3 * DO_ * 2);
    bf16_t* WhhT  = (bf16_t*)alloc((size_t)DO_ * 3 * DO_ * 2);
    bf16_t* W1T   = (bf16_t*)alloc((size_t)DO_ * 4 * DO_ * 2);
    bf16_t* W2T   = (bf16_t*)alloc((size_t)4 * DO_ * DO_ * 2);
    bf16_t* cbbf  = (bf16_t*)alloc((size_t)KC_ * DO_ * 2);
    bf16_t* caWqT = (bf16_t*)alloc((size_t)DF_ * DF_ * 2);
    bf16_t* caWkT = (bf16_t*)alloc((size_t)DO_ * DF_ * 2);
    bf16_t* caWvT = (bf16_t*)alloc((size_t)DO_ * DF_ * 2);
    bf16_t* caWoT = (bf16_t*)alloc((size_t)DF_ * DF_ * 2);

    // ---- once-per-launch prep -------------------------------------------
    auto wt = [&](const float* w, bf16_t* o, int R, int C) {
        int tot = R * C;
        wt_transpose_kernel<<<(tot + 255) / 256, 256, 0, stream>>>(w, o, R, C);
    };
    wt(Wk, WkT, DF_, KVQ_);           wt(Wv, WvT, DF_, KVQ_);
    wt(Wq, WqT, DO_, KVQ_);
    wt(gru_Wih, WihT, KVQ_, 3 * DO_); wt(gru_Whh, WhhT, DO_, 3 * DO_);
    wt(mlp_W1, W1T, DO_, 4 * DO_);    wt(mlp_W2, W2T, 4 * DO_, DO_);
    wt(ca_Wq, caWqT, DF_, DF_);       wt(ca_Wk, caWkT, DO_, DF_);
    wt(ca_Wv, caWvT, DO_, DF_);       wt(ca_Wo, caWoT, DF_, DF_);
    f32_to_bf16_kernel<<<(KC_ * DO_ + 255) / 256, 256, 0, stream>>>(codebook, cbbf, KC_ * DO_);
    cnorm_kernel<<<2, 256, 0, stream>>>(codebook, cn);
    init_acc_kernel<<<1, 32, 0, stream>>>(acc);
    (void)hipMemcpyAsync(inputs, features, sizeof(float) * TOK * DF_, hipMemcpyDeviceToDevice, stream);
    (void)hipMemcpyAsync(slots, conditioning, sizeof(float) * B_ * S_ * DO_, hipMemcpyDeviceToDevice, stream);

    // ---- iterations ------------------------------------------------------
    for (int it = 0; it < ITERS_; ++it) {
        // pos-embed + LN -> kin bf16
        posln_kernel<<<B_ * N_, 256, 0, stream>>>(inputs, positions, Wpos, bpos,
                                                  ln_in_g, ln_in_b, kinbuf);
        // k, v projections (32768 x 256 x 768)
        gemm(stream, kinbuf, 0, 0, DF_, WkT, 0, 0, DF_,
             nullptr, kbuf, 0, 0, KVQ_, (int)TOK, KVQ_, DF_, 1, 1, nullptr, 0);
        gemm(stream, kinbuf, 0, 0, DF_, WvT, 0, 0, DF_,
             nullptr, vbuf, 0, 0, KVQ_, (int)TOK, KVQ_, DF_, 1, 1, nullptr, 0);
        transpose_v_kernel<<<dim3(32, 8, B_), 256, 0, stream>>>(vbuf, vT);
        // q projection
        ln256_kernel<<<512, 256, 0, stream>>>(slots, ln_sl_g, ln_sl_b, snbf);
        gemm(stream, snbf, 0, 0, DO_, WqT, 0, 0, DO_,
             nullptr, qbf, 0, 0, KVQ_, 512, KVQ_, DO_, 1, 1, nullptr, 0);
        // dots: per (b,h): [16 x 1024] = q[16x64] * k[1024x64]^T
        gemm(stream, qbf, (long long)S_ * KVQ_, DPH_, KVQ_,
             kbuf, (long long)N_ * KVQ_, DPH_, KVQ_,
             dots, nullptr, (long long)S_ * H_ * N_, N_, H_ * N_,
             S_, N_, DPH_, B_ * H_, H_, nullptr, 0);
        softmax_sh_kernel<<<B_ * N_ / 256, 256, 0, stream>>>(dots, attnf, out + OUT_ATTN);
        renorm_kernel<<<B_ * S_ * H_, 256, 0, stream>>>(attnf, attnb);
        // updates: per (b,h): [16 x 64] = attn[16x1024] * vT[64x1024]^T
        gemm(stream, attnb, (long long)S_ * H_ * N_, N_, H_ * N_,
             vT, (long long)KVQ_ * N_, (long long)DPH_ * N_, N_,
             nullptr, updates, (long long)S_ * KVQ_, DPH_, KVQ_,
             S_, DPH_, N_, B_ * H_, H_, nullptr, 0);
        // GRU
        f32_to_bf16_kernel<<<512, 256, 0, stream>>>(slots, slotsbf, 512 * 256);
        gemm(stream, updates, 0, 0, KVQ_, WihT, 0, 0, KVQ_,
             gi, nullptr, 0, 0, 3 * DO_, 512, 3 * DO_, KVQ_, 1, 1, gru_bih, GF_BIAS);
        gemm(stream, slotsbf, 0, 0, DO_, WhhT, 0, 0, DO_,
             gh, nullptr, 0, 0, 3 * DO_, 512, 3 * DO_, DO_, 1, 1, gru_bhh, GF_BIAS);
        gru_kernel<<<512, 256, 0, stream>>>(gi, gh, slots);
        // residual MLP
        ln256_kernel<<<512, 256, 0, stream>>>(slots, mlp_ln_g, mlp_ln_b, tbf);
        gemm(stream, tbf, 0, 0, DO_, W1T, 0, 0, DO_,
             nullptr, h1bf, 0, 0, 4 * DO_, 512, 4 * DO_, DO_, 1, 1, mlp_b1, GF_BIAS | GF_RELU);
        gemm(stream, h1bf, 0, 0, 4 * DO_, W2T, 0, 0, 4 * DO_,
             slots, nullptr, 0, 0, DO_, 512, DO_, 4 * DO_, 1, 1, mlp_b2, GF_BIAS | GF_ACCUM);
        // VQ pool
        f32_to_bf16_kernel<<<512, 256, 0, stream>>>(slots, slotsbf, 512 * 256);
        gemm(stream, slotsbf, 0, 0, DO_, cbbf, 0, 0, DO_,
             dp, nullptr, 0, 0, KC_, 512, KC_, DO_, 1, 1, nullptr, 0);
        vq_kernel<<<512, 256, 0, stream>>>(dp, cn, slots, codebook, qsl, indsbuf,
                                           out + OUT_INDS, acc);
        // cross-attention K/V from quantized slots
        gemm(stream, qsl, 0, 0, DO_, caWkT, 0, 0, DO_,
             nullptr, kf, 0, 0, DF_, 512, DF_, DO_, 1, 1, ca_bk, GF_BIAS);
        gemm(stream, qsl, 0, 0, DO_, caWvT, 0, 0, DO_,
             nullptr, vfb, 0, 0, DF_, 512, DF_, DO_, 1, 1, ca_bv, GF_BIAS);
        vft_kernel<<<(B_ * H_ * CAD_ * 32) / 256, 256, 0, stream>>>(vfb, vfT);
        // qf = LN(inputs) @ ca_Wq + bq   (32768 x 768 x 768)
        posln_kernel<<<B_ * N_, 256, 0, stream>>>(inputs, nullptr, nullptr, nullptr,
                                                  ln_ft_g, ln_ft_b, kinbuf);
        gemm(stream, kinbuf, 0, 0, DF_, caWqT, 0, 0, DF_,
             nullptr, qf, 0, 0, DF_, (int)TOK, DF_, DF_, 1, 1, ca_bq, GF_BIAS);
        // logits: per (b,h): [1024 x 16] = qf[1024x192] * kf[16x192]^T
        gemm(stream, qf, (long long)N_ * DF_, CAD_, DF_,
             kf, (long long)S_ * DF_, CAD_, DF_,
             dots, nullptr, (long long)H_ * N_ * S_, (long long)N_ * S_, S_,
             N_, S_, CAD_, B_ * H_, H_, nullptr, 0);
        ca_softmax_kernel<<<B_ * N_ / 256, 256, 0, stream>>>(dots, wbuf, out + OUT_CA);
        // ctx: per (b,h): [1024 x 192] = w[1024x32] * vfT[192x32]^T  (K padded)
        gemm(stream, wbuf, (long long)H_ * N_ * 32, (long long)N_ * 32, 32,
             vfT, (long long)H_ * CAD_ * 32, (long long)CAD_ * 32, 32,
             nullptr, kinbuf, (long long)N_ * DF_, CAD_, DF_,
             N_, CAD_, 32, B_ * H_, H_, nullptr, 0);
        // inputs += ctx @ ca_Wo + bo   (32768 x 768 x 768)
        gemm(stream, kinbuf, 0, 0, DF_, caWoT, 0, 0, DF_,
             inputs, nullptr, 0, 0, DF_, (int)TOK, DF_, DF_, 1, 1, ca_bo,
             GF_BIAS | GF_ACCUM);
    }

    // ---- outputs ---------------------------------------------------------
    (void)hipMemcpyAsync(out + OUT_SLOTS, slots, sizeof(float) * B_ * S_ * DO_,
                         hipMemcpyDeviceToDevice, stream);
    (void)hipMemcpyAsync(out + OUT_INPUTS, inputs, sizeof(float) * TOK * DF_,
                         hipMemcpyDeviceToDevice, stream);
    commit_final_kernel<<<1, 32, 0, stream>>>(acc, out + OUT_COMMIT);
}
